// GINNet_12360915878364
// MI455X (gfx1250) — compile-verified, matrix operand order checked by reading, not run on hardware
//
#include <hip/hip_runtime.h>
#include <hip/hip_bf16.h>

#define D_FEAT 128

typedef __attribute__((ext_vector_type(16))) __bf16 v16bf;
typedef __attribute__((ext_vector_type(8)))  float  v8f;

union FragBF {
    v16bf v;
    unsigned short u[16];
};

__device__ __forceinline__ unsigned short f2b(float f) {
    // let clang lower fptrunc f32->bf16 (RNE); picks HW convert when available
    __bf16 b = (__bf16)f;
    return __builtin_bit_cast(unsigned short, b);
}

// ---------------------------------------------------------------------------
// Convert an f32 matrix to bf16 (weights prep)
// ---------------------------------------------------------------------------
__global__ void __launch_bounds__(256) cvt_bf16(unsigned short* __restrict__ dst,
                                                const float* __restrict__ src, int n) {
    int i = blockIdx.x * 256 + threadIdx.x;
    if (i < n) dst[i] = f2b(src[i]);
}

// ---------------------------------------------------------------------------
// agg = (1 + eps) * h      (eps is a 1-element device buffer)
// ---------------------------------------------------------------------------
__global__ void __launch_bounds__(256) gin_scale(float* __restrict__ agg,
                                                 const float* __restrict__ h,
                                                 const float* __restrict__ epsp,
                                                 int n4) {
    int i = blockIdx.x * 256 + threadIdx.x;
    if (i >= n4) return;
    float s = 1.0f + epsp[0];
    float4 v = ((const float4*)h)[i];
    v.x *= s; v.y *= s; v.z *= s; v.w *= s;
    ((float4*)agg)[i] = v;
}

// ---------------------------------------------------------------------------
// Scatter-add: one wave32 per edge; lane l handles features [4l, 4l+4)
// agg[dst] += h[src]   via hardware f32 atomics (non-returning, STOREcnt)
// ---------------------------------------------------------------------------
__global__ void __launch_bounds__(256) gin_scatter(float* __restrict__ agg,
                                                   const float* __restrict__ h,
                                                   const int* __restrict__ ei,
                                                   int numE) {
    int e    = (int)((blockIdx.x * 256 + threadIdx.x) >> 5);
    int lane = threadIdx.x & 31;
    if (e >= numE) return;
    int src = ei[e];
    int dst = ei[numE + e];
    float4 v = *(const float4*)(h + (size_t)src * D_FEAT + lane * 4);
    float* ap = agg + (size_t)dst * D_FEAT + lane * 4;
    unsafeAtomicAdd(ap + 0, v.x);
    unsafeAtomicAdd(ap + 1, v.y);
    unsafeAtomicAdd(ap + 2, v.z);
    unsafeAtomicAdd(ap + 3, v.w);
}

// ---------------------------------------------------------------------------
// Tensor Data Mover availability (this toolchain: 6-arg builtin)
// ---------------------------------------------------------------------------
#if __has_builtin(__builtin_amdgcn_tensor_load_to_lds) && \
    __has_builtin(__builtin_amdgcn_s_wait_tensorcnt)
#define HAS_TDM 1
typedef __attribute__((ext_vector_type(4))) unsigned tdm_u4;
typedef __attribute__((ext_vector_type(8))) int      tdm_i8;
typedef __attribute__((ext_vector_type(4))) int      tdm_i4;
#else
#define HAS_TDM 0
#endif

// ---------------------------------------------------------------------------
// Fused GIN MLP: hout = relu( relu(z @ w1^T + b1) @ w2^T + b2 )
// Block = 256 threads = 8 waves; block handles 16 rows, wave w handles
// output columns [16w, 16w+16).  bf16 WMMA, f32 accumulate.
// z tile staged into LDS by the Tensor Data Mover (one descriptor per block).
// ---------------------------------------------------------------------------
#define ZPITCH 132  // shorts per row in LDS (padded: 66 dwords, conflict-free)

__global__ void __launch_bounds__(256) gin_mlp(float* __restrict__ hout,
                                               const float* __restrict__ z,
                                               const unsigned short* __restrict__ w1,
                                               const float* __restrict__ b1,
                                               const unsigned short* __restrict__ w2,
                                               const float* __restrict__ b2) {
#if HAS_TDM
    __shared__ float zraw[16 * D_FEAT];          // raw f32 tile (TDM destination)
#endif
    __shared__ unsigned short zb[16 * ZPITCH];   // bf16 A-tile, GEMM1
    __shared__ unsigned short mb[16 * ZPITCH];   // bf16 intermediate, GEMM2

    const int tid  = threadIdx.x;
    const int lane = tid & 31;
    const int wave = tid >> 5;
    const int row0 = blockIdx.x * 16;
    const int n0   = wave * 16;
    const int m    = lane & 15;
    const int half = lane >> 4;

#if HAS_TDM
    // ---- TDM: DMA the 16x128 f32 tile straight into LDS (wave 0 only) ----
    if (wave == 0) {
        unsigned long long ga = (unsigned long long)(z + (size_t)row0 * D_FEAT);
        unsigned lds_off = (unsigned)(unsigned long long)(void*)zraw;  // LDS byte offset
        tdm_u4 g0;
        g0.x = 1u;                                  // count=1, user mode
        g0.y = lds_off;                             // lds_addr
        g0.z = (unsigned)ga;                        // global_addr[31:0]
        g0.w = (unsigned)(ga >> 32) | 0x80000000u;  // global_addr[56:32] | type=2
        tdm_i8 g1;
        g1[0] = 0x00020000;        // workgroup_mask=0, data_size=2 (4 bytes)
        g1[1] = (int)(128u << 16); // tensor_dim0[15:0]=128 in bits[63:48]
        g1[2] = (int)(16u << 16);  // tensor_dim0 hi=0, tensor_dim1[15:0]=16
        g1[3] = (int)(128u << 16); // tensor_dim1 hi=0, tile_dim0=128
        g1[4] = 16;                // tile_dim1=16, tile_dim2=0
        g1[5] = 128;               // tensor_dim0_stride[31:0]=128
        g1[6] = 0;                 // stride0 hi, tensor_dim1_stride lo
        g1[7] = 0;                 // tensor_dim1_stride hi
        tdm_i4 g2 = {0, 0, 0, 0};
        tdm_i4 g3 = {0, 0, 0, 0};
        tdm_i8 g4 = {0, 0, 0, 0, 0, 0, 0, 0};
        __builtin_amdgcn_tensor_load_to_lds(g0, g1, g2, g3, g4, 0);
        __builtin_amdgcn_s_wait_tensorcnt(0);
    }
    __syncthreads();
    // ---- convert f32 tile (LDS) -> bf16 A-tile (LDS) ----
    for (int i = tid; i < 16 * 32; i += 256) {
        int r  = i >> 5;
        int c4 = i & 31;
        float4 v = *(const float4*)(zraw + r * D_FEAT + c4 * 4);
        unsigned short* d = zb + r * ZPITCH + c4 * 4;
        d[0] = f2b(v.x); d[1] = f2b(v.y); d[2] = f2b(v.z); d[3] = f2b(v.w);
    }
#else
    // ---- fallback: stage z tile (16 x 128 f32 -> bf16 LDS) via VGPRs ----
    for (int i = tid; i < 16 * 32; i += 256) {
        int r  = i >> 5;
        int c4 = i & 31;
        float4 v = *(const float4*)(z + (size_t)(row0 + r) * D_FEAT + c4 * 4);
        unsigned short* d = zb + r * ZPITCH + c4 * 4;
        d[0] = f2b(v.x); d[1] = f2b(v.y); d[2] = f2b(v.z); d[3] = f2b(v.w);
    }
#endif
    __syncthreads();

    // ---- GEMM1: acc[16x16] = z(16x128) x w1^T(128x16) ----
    v8f acc = {0.f, 0.f, 0.f, 0.f, 0.f, 0.f, 0.f, 0.f};
    for (int c = 0; c < 4; ++c) {
        FragBF a, b;
#pragma unroll
        for (int p = 0; p < 8; ++p) {
            // ISA 16-bit A layout: VGPR p holds K pair; lanes 16-31 shift K by 8
            int k0 = 32 * c + ((p < 4) ? 2 * p : 16 + 2 * (p - 4)) + 8 * half;
            a.u[2 * p]     = zb[m * ZPITCH + k0];
            a.u[2 * p + 1] = zb[m * ZPITCH + k0 + 1];
            unsigned wp = *(const unsigned*)(w1 + (size_t)(n0 + m) * D_FEAT + k0);
            b.u[2 * p]     = (unsigned short)(wp & 0xFFFFu);
            b.u[2 * p + 1] = (unsigned short)(wp >> 16);
        }
        acc = __builtin_amdgcn_wmma_f32_16x16x32_bf16(false, a.v, false, b.v,
                                                      (short)0, acc, false, false);
    }

    // bias + ReLU, write intermediate as bf16 A-tile for GEMM2
    {
        float bias = b1[n0 + m];
#pragma unroll
        for (int r = 0; r < 8; ++r) {
            float v = acc[r] + bias;           // C layout: M = r + 8*half, N = m
            v = v > 0.f ? v : 0.f;
            mb[(r + 8 * half) * ZPITCH + n0 + m] = f2b(v);
        }
    }
    __syncthreads();

    // ---- GEMM2: acc2 = mid(16x128) x w2^T(128x16) ----
    v8f acc2 = {0.f, 0.f, 0.f, 0.f, 0.f, 0.f, 0.f, 0.f};
    for (int c = 0; c < 4; ++c) {
        FragBF a, b;
#pragma unroll
        for (int p = 0; p < 8; ++p) {
            int k0 = 32 * c + ((p < 4) ? 2 * p : 16 + 2 * (p - 4)) + 8 * half;
            a.u[2 * p]     = mb[m * ZPITCH + k0];
            a.u[2 * p + 1] = mb[m * ZPITCH + k0 + 1];
            unsigned wp = *(const unsigned*)(w2 + (size_t)(n0 + m) * D_FEAT + k0);
            b.u[2 * p]     = (unsigned short)(wp & 0xFFFFu);
            b.u[2 * p + 1] = (unsigned short)(wp >> 16);
        }
        acc2 = __builtin_amdgcn_wmma_f32_16x16x32_bf16(false, a.v, false, b.v,
                                                       (short)0, acc2, false, false);
    }

    // bias + outer ReLU, write f32 output
    {
        float bias = b2[n0 + m];
#pragma unroll
        for (int r = 0; r < 8; ++r) {
            float v = acc2[r] + bias;
            v = v > 0.f ? v : 0.f;
            hout[(size_t)(row0 + r + 8 * half) * D_FEAT + n0 + m] = v;
        }
    }
}

// ---------------------------------------------------------------------------
// Head: out = sigmoid( relu(h @ fc1^T + fb1) @ fc2^T + fb2 )
// one block of 64 threads per node (thread j = hidden unit j)
// ---------------------------------------------------------------------------
__global__ void __launch_bounds__(64) gin_head(float* __restrict__ out,
                                               const float* __restrict__ h,
                                               const float* __restrict__ fw1,
                                               const float* __restrict__ fb1,
                                               const float* __restrict__ fw2,
                                               const float* __restrict__ fb2) {
    __shared__ float red[64];
    const int node = blockIdx.x;
    const int j    = threadIdx.x;

    const float4* hp = (const float4*)(h + (size_t)node * D_FEAT);
    const float4* wp = (const float4*)(fw1 + (size_t)j * D_FEAT);
    float acc = 0.f;
#pragma unroll 4
    for (int k = 0; k < 32; ++k) {
        float4 a = hp[k];
        float4 b = wp[k];
        acc += a.x * b.x + a.y * b.y + a.z * b.z + a.w * b.w;
    }
    float t = acc + fb1[j];
    t = t > 0.f ? t : 0.f;
    red[j] = t * fw2[j];
    __syncthreads();
    for (int s = 32; s > 0; s >>= 1) {
        if (j < s) red[j] += red[j + s];
        __syncthreads();
    }
    if (j == 0) {
        float zf = red[0] + fb2[0];
        out[node] = 1.f / (1.f + __expf(-zf));
    }
}

// ---------------------------------------------------------------------------
extern "C" void kernel_launch(void* const* d_in, const int* in_sizes, int n_in,
                              void* d_out, int out_size, void* d_ws, size_t ws_size,
                              hipStream_t stream) {
    const float* x  = (const float*)d_in[0];
    const int*   ei = (const int*)d_in[1];
    const int N = in_sizes[0] / D_FEAT;
    const int E = in_sizes[1] / 2;

    const float* cw1[3]  = { (const float*)d_in[2],  (const float*)d_in[7],  (const float*)d_in[12] };
    const float* cb1[3]  = { (const float*)d_in[3],  (const float*)d_in[8],  (const float*)d_in[13] };
    const float* cw2[3]  = { (const float*)d_in[4],  (const float*)d_in[9],  (const float*)d_in[14] };
    const float* cb2[3]  = { (const float*)d_in[5],  (const float*)d_in[10], (const float*)d_in[15] };
    const float* ceps[3] = { (const float*)d_in[6],  (const float*)d_in[11], (const float*)d_in[16] };
    const float* fw1 = (const float*)d_in[17];
    const float* fb1 = (const float*)d_in[18];
    const float* fw2 = (const float*)d_in[19];
    const float* fb2 = (const float*)d_in[20];

    char* ws = (char*)d_ws;
    const size_t hbytes = (size_t)N * D_FEAT * sizeof(float);
    float* h   = (float*)ws;                        // node features between layers
    float* agg = (float*)(ws + hbytes);             // (1+eps)h + sum_neighbors
    unsigned short* wb = (unsigned short*)(ws + 2 * hbytes);  // bf16 weights
    const int WM = D_FEAT * D_FEAT;                 // 16384 elements per matrix

    // prep: convert 6 conv weight matrices to bf16
    for (int l = 0; l < 3; ++l) {
        cvt_bf16<<<(WM + 255) / 256, 256, 0, stream>>>(wb + (size_t)(2 * l) * WM,     cw1[l], WM);
        cvt_bf16<<<(WM + 255) / 256, 256, 0, stream>>>(wb + (size_t)(2 * l + 1) * WM, cw2[l], WM);
    }

    const int n4 = N * (D_FEAT / 4);
    const float* hcur = x;
    for (int l = 0; l < 3; ++l) {
        gin_scale<<<(n4 + 255) / 256, 256, 0, stream>>>(agg, hcur, ceps[l], n4);
        gin_scatter<<<(E + 7) / 8, 256, 0, stream>>>(agg, hcur, ei, E);
        gin_mlp<<<N / 16, 256, 0, stream>>>(h, agg,
                                            wb + (size_t)(2 * l) * WM,     cb1[l],
                                            wb + (size_t)(2 * l + 1) * WM, cb2[l]);
        hcur = h;
    }

    gin_head<<<N, 64, 0, stream>>>((float*)d_out, h, fw1, fb1, fw2, fb2);
}
